// AFF_61323543052519
// MI455X (gfx1250) — compile-verified, matrix operand order checked by reading, not run on hardware
//
#include <hip/hip_runtime.h>
#include <hip/hip_bf16.h>
#include <stdint.h>

// ---------------------------------------------------------------------------
// Continuous-conv network (Open3D-style) for MI455X / gfx1250.
//   layer1: scatter(concat(x,y)) -> acc(N,64,64) -> GEMM(4096x64) -> BN+ReLU
//   layer2: scatter(bnrelu(h1))  -> acc(N,64,64) -> GEMM(4096x32) -> BN+sigmoid
//   out = 2*x*wei + 2*y*(1-wei)
// Biases cancel inside BatchNorm -> skipped.
// acc kept in bf16 (GLOBAL_ATOMIC_PK_ADD_BF16), GEMM via v_wmma_f32_16x16x32_bf16
// with B staged through LDS using GLOBAL_LOAD_ASYNC_TO_LDS_B128 (double buffer,
// 4 k-steps per chunk). GEMM main loop is branch-free (no exec-mask churn).
// ---------------------------------------------------------------------------

#define N_PTS 30000
#define NCH   32
#define CIN   64
#define KBINS 64
#define KROW  4096        // KBINS * CIN
#define E_PADN 1200000

typedef __attribute__((ext_vector_type(16))) __bf16 v16bf;
typedef __attribute__((ext_vector_type(8)))  float  v8f;
typedef __attribute__((address_space(3))) __hip_bfloat16 lds_bf16;

union frag16 { uint4 u[2]; v16bf v; };

__device__ __forceinline__ unsigned short f2bf_rn(float f) {
  union { float f; uint32_t u; } v; v.f = f;
  uint32_t u = v.u;
  u += 0x7fffu + ((u >> 16) & 1u);      // round to nearest even
  return (unsigned short)(u >> 16);
}

template <int Nn>
__device__ __forceinline__ void wait_asynccnt() {
#if __has_builtin(__builtin_amdgcn_s_wait_asynccnt)
  __builtin_amdgcn_s_wait_asynccnt(Nn);
#else
  asm volatile("s_wait_asynccnt %0" :: "n"(Nn));
#endif
}

// --------------------------- edge geometry ---------------------------------
__global__ void edge_geom(const float* __restrict__ pos,
                          const int*  __restrict__ esrc,
                          const int*  __restrict__ edst,
                          const int*  __restrict__ emask,
                          float* __restrict__ w8,
                          int*   __restrict__ kbase) {
  int e = blockIdx.x * blockDim.x + threadIdx.x;
  if (e >= E_PADN) return;
  const float RAD = (1.5f * 6.0f * 0.025f) * 0.5f;
  int s = esrc[e], d = edst[e];
  float inv = 1.0f / RAD;
  float vx = (pos[s * 3 + 0] - pos[d * 3 + 0]) * inv;
  float vy = (pos[s * 3 + 1] - pos[d * 3 + 1]) * inv;
  float vz = (pos[s * 3 + 2] - pos[d * 3 + 2]) * inv;
  float r2 = vx * vx + vy * vy + vz * vz;
  float t1 = 1.0f - r2;
  float win = fminf(fmaxf(t1 * t1 * t1, 0.0f), 1.0f);
  win *= (emask[e] != 0) ? 1.0f : 0.0f;
  float l2 = sqrtf(fmaxf(r2, 1e-12f));
  float linf = fmaxf(fabsf(vx), fmaxf(fabsf(vy), fabsf(vz)));
  float scale = (linf > 0.0f) ? (l2 / fmaxf(linf, 1e-12f)) : 0.0f;
  float gx = (vx * scale + 1.0f) * 1.5f;
  float gy = (vy * scale + 1.0f) * 1.5f;
  float gz = (vz * scale + 1.0f) * 1.5f;
  int ix = (int)fminf(fmaxf(floorf(gx), 0.0f), 2.0f);
  int iy = (int)fminf(fmaxf(floorf(gy), 0.0f), 2.0f);
  int iz = (int)fminf(fmaxf(floorf(gz), 0.0f), 2.0f);
  float fx = gx - (float)ix, fy = gy - (float)iy, fz = gz - (float)iz;
  kbase[e] = ix * 16 + iy * 4 + iz;
  float wz[2] = {1.0f - fx, fx};
  float wy[2] = {1.0f - fy, fy};
  float wx[2] = {1.0f - fz, fz};
  #pragma unroll
  for (int dz = 0; dz < 2; ++dz)
    #pragma unroll
    for (int dy = 0; dy < 2; ++dy)
      #pragma unroll
      for (int dx = 0; dx < 2; ++dx)
        w8[(size_t)e * 8 + dz * 4 + dy * 2 + dx] = win * wz[dz] * wy[dy] * wx[dx];
}

// --------------------------- edge scatter ----------------------------------
// One wave per edge; lane p handles channel pair (2p, 2p+1). Edge-uniform data
// is scalarized via readfirstlane (scalar loads + scalar zero-skip branch).
__global__ void edge_scatter(const float* __restrict__ fx,
                             const float* __restrict__ fy,
                             const float* __restrict__ h,
                             const float* __restrict__ scl,
                             const float* __restrict__ shf,
                             int mode,
                             const int* __restrict__ esrc,
                             const int* __restrict__ edst,
                             const float* __restrict__ w8,
                             const int*  __restrict__ kbase,
                             __hip_bfloat16* __restrict__ acc) {
  int tid = blockIdx.x * blockDim.x + threadIdx.x;
  int e = __builtin_amdgcn_readfirstlane(tid >> 5);   // wave-uniform edge id
  int p = threadIdx.x & 31;
  if (e >= E_PADN) return;
  int s  = __builtin_amdgcn_readfirstlane(esrc[e]);
  int d  = __builtin_amdgcn_readfirstlane(edst[e]);
  int kb = __builtin_amdgcn_readfirstlane(kbase[e]);
  int ch = 2 * p;
  float f0, f1;
  if (mode == 0) {
    const float* base = (ch < NCH) ? fx : fy;
    int c = (ch < NCH) ? ch : (ch - NCH);
    f0 = base[(size_t)s * NCH + c];
    f1 = base[(size_t)s * NCH + c + 1];
  } else {
    size_t idx = (size_t)s * CIN + ch;
    f0 = fmaxf(0.0f, h[idx]     * scl[ch]     + shf[ch]);
    f1 = fmaxf(0.0f, h[idx + 1] * scl[ch + 1] + shf[ch + 1]);
  }
  size_t rowbase = ((size_t)d * KBINS + kb) * CIN + ch;
  const int coff[8] = {0, 1, 4, 5, 16, 17, 20, 21};
  #pragma unroll
  for (int c = 0; c < 8; ++c) {
    float w = __builtin_amdgcn_readfirstlane(w8[(size_t)e * 8 + c]);
    if (w == 0.0f) continue;
    uint32_t pk = (uint32_t)f2bf_rn(w * f0) | ((uint32_t)f2bf_rn(w * f1) << 16);
    __hip_bfloat16* pp = acc + rowbase + (size_t)coff[c] * CIN;
    asm volatile("global_atomic_pk_add_bf16 %0, %1, off"
                 :: "v"(pp), "v"(pk) : "memory");
  }
}

// --------------------------- weight pre-swizzle ----------------------------
// Fragment (kk, t): 32 lanes x 16 bf16; lane -> output column, K pattern
// mirrors the documented 16-bit WMMA A layout (lane->N here).
__global__ void pack_w(const float* __restrict__ W,
                       __hip_bfloat16* __restrict__ dst, int cout) {
  int wpt = cout >> 4;
  int tid = blockIdx.x * blockDim.x + threadIdx.x;
  int total = (KROW / 32) * wpt * 512;
  if (tid >= total) return;
  int j    = tid & 15;
  int lane = (tid >> 4) & 31;
  int frag = tid >> 9;
  int kk = frag / wpt;
  int t  = frag % wpt;
  int hi = lane >> 4;
  int krel = (j < 8) ? (hi * 8 + j) : (16 + hi * 8 + (j - 8));
  int k = kk * 32 + krel;
  int col = t * 16 + (lane & 15);
  dst[tid] = __hip_bfloat16(W[(size_t)k * cout + col]);
}

// --------------------------- WMMA GEMM -------------------------------------
// Block = 128 threads (4 waves); each wave owns one 16-node tile and ALL
// column tiles (WPT accumulators). A (the big 245MB stream) is loaded exactly
// once, two b128 loads per lane per k-step. B is staged in LDS chunks of
// KS=4 k-steps via GLOBAL_LOAD_ASYNC_TO_LDS_B128 (double buffer), shared by
// all 4 waves. Main loop is branch-free; tail waves compute on tile 0 and
// simply skip the epilogue (so no exec-mask save/restore churn).
template <int COUT>
__global__ __launch_bounds__(128)
void wmma_gemm(const __hip_bfloat16* __restrict__ acc,
               const __hip_bfloat16* __restrict__ wpack,
               float* __restrict__ hout,
               float* __restrict__ ssum,
               float* __restrict__ ssq) {
  constexpr int WPT   = COUT / 16;          // column tiles (4 or 2)
  constexpr int KS    = 4;                  // k-steps per staged chunk
  constexpr int CHUNK = KS * WPT * 512;     // bf16 elements per chunk
  constexpr int G     = (CHUNK * 2 / 16) / 128;  // b128 units per thread
  constexpr int NCC   = (KROW / 32) / KS;   // 32 chunks

  __shared__ __align__(16) __hip_bfloat16 sB[2 * CHUNK];

  int lane = threadIdx.x & 31;
  int wave = threadIdx.x >> 5;
  int tile = blockIdx.x * 4 + wave;
  bool active = tile < (N_PTS / 16);
  int node0 = (active ? tile : 0) << 4;     // tail waves re-run tile 0 (valid)

  // A-fragment base: lane<16 -> row=lane, K halves {0..7,16..23};
  //                  lane>=16 -> row=lane-16, K halves {8..15,24..31}.
  const __hip_bfloat16* ap =
      acc + (size_t)(node0 + (lane & 15)) * KROW + ((lane >> 4) << 3);

  // async stage of B chunk `cc` into LDS buffer `buf`
  auto stage = [&](int cc, int buf) {
    const __hip_bfloat16* gsrc = wpack + (size_t)cc * CHUNK;
    #pragma unroll
    for (int g = 0; g < G; ++g) {
      int iu = threadIdx.x + g * 128;              // 16B unit index
      const __hip_bfloat16* gp = gsrc + iu * 8;
      lds_bf16* lp = (lds_bf16*)(sB + buf * CHUNK + iu * 8);
      asm volatile("global_load_async_to_lds_b128 %0, %1, off"
                   :: "v"(lp), "v"(gp) : "memory");
    }
  };

  v8f c[WPT];
  #pragma unroll
  for (int t = 0; t < WPT; ++t) c[t] = v8f{};

  stage(0, 0);
  for (int cc = 0; cc < NCC; ++cc) {
    if (cc + 1 < NCC) { stage(cc + 1, (cc + 1) & 1); wait_asynccnt<G>(); }
    else              { wait_asynccnt<0>(); }
    __syncthreads();                    // chunk cc fully in LDS for all waves
    const __hip_bfloat16* sp = sB + (cc & 1) * CHUNK + lane * 16;
    #pragma unroll
    for (int ks = 0; ks < KS; ++ks) {
      frag16 a;
      a.u[0] = *(const uint4*)(ap + ks * 32);
      a.u[1] = *(const uint4*)(ap + ks * 32 + 16);
      frag16 b[WPT];
      #pragma unroll
      for (int t = 0; t < WPT; ++t) {
        b[t].u[0] = *(const uint4*)(sp + (ks * WPT + t) * 512);
        b[t].u[1] = *(const uint4*)(sp + (ks * WPT + t) * 512 + 8);
      }
      #pragma unroll
      for (int t = 0; t < WPT; ++t)
        c[t] = __builtin_amdgcn_wmma_f32_16x16x32_bf16(
                   false, a.v, false, b[t].v, (short)0, c[t], false, false);
    }
    ap += KS * 32;
    __syncthreads();                    // done reading before chunk cc+2 lands
  }

  if (!active) return;
  // D layout: lane -> col (lane&15), VGPR r -> row r (+8 for lanes 16..31).
  int hi = lane >> 4;
  #pragma unroll
  for (int t = 0; t < WPT; ++t) {
    int col = t * 16 + (lane & 15);
    float s = 0.0f, s2 = 0.0f;
    #pragma unroll
    for (int r = 0; r < 8; ++r) {
      float val = c[t][r];
      hout[(size_t)(node0 + r + hi * 8) * COUT + col] = val;
      s += val;
      s2 += val * val;
    }
    s  += __shfl_xor(s, 16, 32);
    s2 += __shfl_xor(s2, 16, 32);
    if (lane < 16) {
      atomicAdd(&ssum[col], s);
      atomicAdd(&ssq[col], s2);
    }
  }
}

// --------------------------- BN finalize -----------------------------------
__global__ void bn_finalize(const float* __restrict__ sum,
                            const float* __restrict__ sq,
                            const float* __restrict__ gamma,
                            const float* __restrict__ beta,
                            float* __restrict__ scale,
                            float* __restrict__ shift, int C) {
  int c = threadIdx.x;
  if (c >= C) return;
  float inv = 1.0f / (float)N_PTS;
  float m = sum[c] * inv;
  float var = fmaxf(sq[c] * inv - m * m, 0.0f);
  float rstd = rsqrtf(var + 1e-5f);
  float sc = gamma[c] * rstd;
  scale[c] = sc;
  shift[c] = beta[c] - m * sc;
}

// --------------------------- final blend -----------------------------------
__global__ void final_blend(const float* __restrict__ x,
                            const float* __restrict__ y,
                            const float* __restrict__ h2,
                            const float* __restrict__ scale,
                            const float* __restrict__ shift,
                            float* __restrict__ out) {
  int tid = blockIdx.x * blockDim.x + threadIdx.x;
  if (tid >= N_PTS * NCH) return;
  int ch = tid & (NCH - 1);
  float z = h2[tid] * scale[ch] + shift[ch];
  float wei = 1.0f / (1.0f + __expf(-z));
  out[tid] = 2.0f * x[tid] * wei + 2.0f * y[tid] * (1.0f - wei);
}

// ---------------------------------------------------------------------------
extern "C" void kernel_launch(void* const* d_in, const int* in_sizes, int n_in,
                              void* d_out, int out_size, void* d_ws, size_t ws_size,
                              hipStream_t stream) {
  const float* x   = (const float*)d_in[0];
  const float* y   = (const float*)d_in[1];
  const float* pos = (const float*)d_in[2];
  const float* k1  = (const float*)d_in[3];
  const float* g1  = (const float*)d_in[5];
  const float* b1  = (const float*)d_in[6];
  const float* k2  = (const float*)d_in[7];
  const float* g2  = (const float*)d_in[9];
  const float* b2  = (const float*)d_in[10];
  const int* esrc  = (const int*)d_in[11];
  const int* edst  = (const int*)d_in[12];
  const int* emask = (const int*)d_in[13];
  float* out = (float*)d_out;

  char* ws = (char*)d_ws;
  size_t o = 0;
  auto wsalloc = [&](size_t bytes) -> void* {
    void* p = ws + o;
    o = (o + bytes + 255) & ~(size_t)255;
    return p;
  };
  __hip_bfloat16* accB = (__hip_bfloat16*)wsalloc((size_t)N_PTS * KROW * 2);
  float* h1 = (float*)wsalloc((size_t)N_PTS * 64 * 4);
  float* h2 = (float*)wsalloc((size_t)N_PTS * 32 * 4);
  __hip_bfloat16* wp1 = (__hip_bfloat16*)wsalloc((size_t)KROW * 64 * 2);
  __hip_bfloat16* wp2 = (__hip_bfloat16*)wsalloc((size_t)KROW * 32 * 2);
  float* w8 = (float*)wsalloc((size_t)E_PADN * 8 * 4);
  int*   kb = (int*)wsalloc((size_t)E_PADN * 4);
  float* stats = (float*)wsalloc(2048);
  float *sum1 = stats,       *sq1 = stats + 64,  *scl1 = stats + 128, *shf1 = stats + 192;
  float *sum2 = stats + 256, *sq2 = stats + 320, *scl2 = stats + 384, *shf2 = stats + 448;

  hipMemsetAsync(accB, 0, (size_t)N_PTS * KROW * 2, stream);
  hipMemsetAsync(stats, 0, 2048, stream);

  pack_w<<<(128 * 4 * 512 + 255) / 256, 256, 0, stream>>>(k1, wp1, 64);
  pack_w<<<(128 * 2 * 512 + 255) / 256, 256, 0, stream>>>(k2, wp2, 32);
  edge_geom<<<(E_PADN + 255) / 256, 256, 0, stream>>>(pos, esrc, edst, emask, w8, kb);

  const int GEMM_GRID = (N_PTS / 16 + 3) / 4;   // 469 blocks, 4 node-tiles each

  // layer 1
  edge_scatter<<<E_PADN * 32 / 256, 256, 0, stream>>>(
      x, y, nullptr, nullptr, nullptr, 0, esrc, edst, w8, kb, accB);
  wmma_gemm<64><<<GEMM_GRID, 128, 0, stream>>>(accB, wp1, h1, sum1, sq1);
  bn_finalize<<<1, 64, 0, stream>>>(sum1, sq1, g1, b1, scl1, shf1, 64);

  // layer 2
  hipMemsetAsync(accB, 0, (size_t)N_PTS * KROW * 2, stream);
  edge_scatter<<<E_PADN * 32 / 256, 256, 0, stream>>>(
      nullptr, nullptr, h1, scl1, shf1, 1, esrc, edst, w8, kb, accB);
  wmma_gemm<32><<<GEMM_GRID, 128, 0, stream>>>(accB, wp2, h2, sum2, sq2);
  bn_finalize<<<1, 64, 0, stream>>>(sum2, sq2, g2, b2, scl2, shf2, 32);

  final_blend<<<(N_PTS * NCH + 255) / 256, 256, 0, stream>>>(
      x, y, h2, scl2, shf2, out);
}
